// FeaStNet_18640158065032
// MI455X (gfx1250) — matrix operand order repaired and checked by reading, not compiled, not run on hardware
//
#include <hip/hip_runtime.h>
#include <hip/hip_bf16.h>

// ---------------------------------------------------------------------------
// FeaStNet forward on MI455X (gfx1250, wave32, WMMA).
//   Dense layers (M = h@W per conv, lin2) use v_wmma_f32_16x16x32_f16 with
//   pre-packed f16 fragment buffers so the hot loop is 2x 32B (v16h) load + wmma,
//   branch-free. Edge pass: one wave32 per edge, coalesced gathers from the
//   node-level M buffer, f32 global atomics for scatter-mean. Loss via
//   two-level reduction.
// ---------------------------------------------------------------------------

typedef __attribute__((ext_vector_type(16))) _Float16 v16h;
typedef __attribute__((ext_vector_type(8)))  float    v8f;

// ---------------------------------------------------------------------------
// Fragment packing.
// A-matrix 16x32 f16 VGPR layout (ISA 7.12.2): lane-half h holds, per row,
// k = {0..7,16..23} (h=0) / {8..15,24..31} (h=1). We store each row's 32-k
// chunk linearly (j == k-within-chunk) so half h reads the
// contiguous 16 halfs at offset 16*h, i.e. k = 16h..16h+15, the same order as B.
// ---------------------------------------------------------------------------
__global__ void pack_a_f16(const float* __restrict__ hin, _Float16* __restrict__ Ap,
                           int N, int cin, int nrowsPad, int nchunks) {
    long long tot = (long long)nrowsPad * nchunks * 32;
    long long i = (long long)blockIdx.x * blockDim.x + threadIdx.x;
    if (i >= tot) return;
    int j = (int)(i & 31);
    long long rc = i >> 5;
    int c   = (int)(rc % nchunks);
    int row = (int)(rc / nchunks);
    int g = j >> 3;
    int off = 0 * g;
    int k = c * 32 + j + off;
    float v = (row < N && k < cin) ? hin[(long long)row * cin + k] : 0.0f;
    Ap[i] = (_Float16)v;
}

// B 32x16 f16 fragment (contiguous k order, as for A): lane-half h holds k = {0..15}+16h for its column;
// linear in k, so store [chunk][col][32] with j == k-within-chunk.
__global__ void pack_b_f16(const float* __restrict__ W, _Float16* __restrict__ Bp,
                           int cin, int kout, int nchunks) {
    long long tot = (long long)nchunks * kout * 32;
    long long i = (long long)blockIdx.x * blockDim.x + threadIdx.x;
    if (i >= tot) return;
    int j = (int)(i & 31);
    long long cc = i >> 5;
    int col = (int)(cc % kout);
    int c   = (int)(cc / kout);
    int k = c * 32 + j;
    float v = (k < cin) ? W[(long long)k * kout + col] : 0.0f;
    Bp[i] = (_Float16)v;
}

// ---------------------------------------------------------------------------
// WMMA GEMM on packed fragments: C[nrows,kout] = A[nrows,K] * B[K,kout].
// One wave32 per 16x16 tile. Hot loop: one v16h (32B) A load + one v16h B
// load + one v_wmma_f32_16x16x32_f16. Optional fused bias + ReLU.
// ---------------------------------------------------------------------------
__global__ void gemm_wmma_packed(const _Float16* __restrict__ Ap, // [nrowsPad][nchunks][32]
                                 const _Float16* __restrict__ Bp, // [nchunks][kout][32]
                                 float* __restrict__ Cm,          // [nrows][kout]
                                 const float* __restrict__ bias,  // nullable [kout]
                                 int nrows, int nrowsPad, int nchunks, int kout,
                                 int fuse_relu) {
    const int lane = threadIdx.x & 31;                 // wave32
    const int wgid = blockIdx.x * (blockDim.x >> 5) + (threadIdx.x >> 5);
    const int tilesN = kout >> 4;
    const int tilesM = nrowsPad >> 4;
    if (wgid >= tilesM * tilesN) return;               // wave-uniform: EXEC stays all-1
    const int mt = wgid / tilesN;
    const int nt = wgid % tilesN;
    const int half = lane >> 4;
    const int idx  = lane & 15;
    const int arow = mt * 16 + idx;                    // row this lane feeds
    const int bcol = nt * 16 + idx;                    // col this lane feeds / C col

    const _Float16* Aptr = Ap + ((long long)arow * nchunks) * 32 + 16 * half;
    const _Float16* Bptr = Bp + ((long long)bcol) * 32 + 16 * half;
    const long long bstride = (long long)kout * 32;

    v8f acc = {};
    for (int c = 0; c < nchunks; ++c) {
        v16h af = *(const v16h*)(Aptr + (long long)c * 32);
        v16h bf = *(const v16h*)(Bptr + (long long)c * bstride);
        acc = __builtin_amdgcn_wmma_f32_16x16x32_f16(
            /*neg_a=*/false, af, /*neg_b=*/false, bf,
            /*c_mod=*/(short)0, acc, /*reuse_a=*/false, /*reuse_b=*/false);
    }

#pragma unroll
    for (int r = 0; r < 8; ++r) {
        int orow = mt * 16 + r + 8 * half;             // C/D: VGPR r -> row r+8*half
        if (orow < nrows) {
            float v = acc[r];
            if (bias) v += bias[bcol];
            if (fuse_relu) v = fmaxf(v, 0.0f);
            Cm[(long long)orow * kout + bcol] = v;
        }
    }
}

// ---------------------------------------------------------------------------
// Small elementwise / setup kernels
// ---------------------------------------------------------------------------
__global__ void fill_f32(float* __restrict__ p, float v, long long n) {
    long long i = (long long)blockIdx.x * blockDim.x + threadIdx.x;
    if (i < n) p[i] = v;
}

__global__ void degree_kernel(const int* __restrict__ dst, int E, float* __restrict__ cnt) {
    int e = blockIdx.x * blockDim.x + threadIdx.x;
    if (e < E) atomicAdd(&cnt[dst[e]], 1.0f);
}

__global__ void lin1_kernel(const float* __restrict__ x, const float* __restrict__ W,
                            const float* __restrict__ b, float* __restrict__ h, int N) {
    int n = blockIdx.x * blockDim.x + threadIdx.x;
    if (n >= N) return;
    float x0 = x[n * 3 + 0], x1 = x[n * 3 + 1], x2 = x[n * 3 + 2];
#pragma unroll
    for (int j = 0; j < 16; ++j) {
        float v = x0 * W[j] + x1 * W[16 + j] + x2 * W[32 + j] + b[j];
        h[n * 16 + j] = fmaxf(v, 0.0f);
    }
}

// Per-node attention pre-logits: A[n,h] = sum_c hin[n,c] * u[c,h]
__global__ void attn_pre(const float* __restrict__ hin, const float* __restrict__ u,
                         float* __restrict__ A, int N, int cin) {
    int n = blockIdx.x * blockDim.x + threadIdx.x;
    if (n >= N) return;
    float acc[8] = {0, 0, 0, 0, 0, 0, 0, 0};
    const float* hr = hin + (long long)n * cin;
    for (int c = 0; c < cin; ++c) {
        float hv = hr[c];
#pragma unroll
        for (int h = 0; h < 8; ++h) acc[h] += hv * u[c * 8 + h];
    }
#pragma unroll
    for (int h = 0; h < 8; ++h) A[n * 8 + h] = acc[h];
}

// ---------------------------------------------------------------------------
// Edge pass: one wave32 per edge (self loops at index >= E).
// ---------------------------------------------------------------------------
__global__ void feast_edge_kernel(const int* __restrict__ ei, int E, int N,
                                  const float* __restrict__ A,    // [N,8]
                                  const float* __restrict__ cvec, // [8]
                                  const float* __restrict__ M,    // [N, 8*cout]
                                  float* __restrict__ agg,        // [N, cout] pre-zeroed
                                  int cout) {
    int lane = threadIdx.x & 31;
    int wid  = (int)(((long long)blockIdx.x * blockDim.x + threadIdx.x) >> 5);
    int ET = E + N;
    if (wid >= ET) return;                            // wave-uniform
    int s, d;
    if (wid < E) { s = ei[wid]; d = ei[E + wid]; }
    else         { s = d = wid - E; }                 // self loop

    const int K = cout * 8;
    const float* Ms = M + (long long)s * K;
    __builtin_prefetch(Ms + lane * 16, 0, 0);         // global_prefetch_b8

    float l[8];
    float mx = -3.0e38f;
#pragma unroll
    for (int h = 0; h < 8; ++h) {
        l[h] = A[s * 8 + h] - A[d * 8 + h] + cvec[h];
        mx = fmaxf(mx, l[h]);
    }
    float sum = 0.0f;
#pragma unroll
    for (int h = 0; h < 8; ++h) { l[h] = __expf(l[h] - mx); sum += l[h]; }
    float inv = 1.0f / sum;

    float* Ad = agg + (long long)d * cout;
    for (int o = lane; o < cout; o += 32) {
        float acc = 0.0f;
#pragma unroll
        for (int h = 0; h < 8; ++h) acc += l[h] * Ms[h * cout + o];
        atomicAdd(&Ad[o], acc * inv);
    }
}

// mean aggregation + bias + ReLU
__global__ void finish_kernel(float* __restrict__ hout, const float* __restrict__ cnt,
                              const float* __restrict__ b, int N, int cout) {
    long long i = (long long)blockIdx.x * blockDim.x + threadIdx.x;
    if (i >= (long long)N * cout) return;
    int n = (int)(i / cout), o = (int)(i % cout);
    hout[i] = fmaxf(hout[i] / cnt[n] + b[o], 0.0f);
}

// lin3 + sigmoid: one wave32 per node, 1024-deep dot via lane-strided loads.
__global__ void lin3_sigmoid(const float* __restrict__ h, const float* __restrict__ w,
                             const float* __restrict__ b, float* __restrict__ p, int N) {
    int lane = threadIdx.x & 31;
    int n = (int)(((long long)blockIdx.x * blockDim.x + threadIdx.x) >> 5);
    if (n >= N) return;
    const float* hr = h + (long long)n * 1024;
    float acc = 0.0f;
    for (int k = lane; k < 1024; k += 32) acc += hr[k] * w[k];
#pragma unroll
    for (int off = 16; off > 0; off >>= 1) acc += __shfl_down(acc, off, 32);
    if (lane == 0) p[n] = 1.0f / (1.0f + __expf(-(acc + b[0])));
}

// ---------------------------------------------------------------------------
// Loss: loss = sum_pos(bce)/(2*pos) + sum_neg(bce)/(2*neg)  == mean(w*bce).
// ---------------------------------------------------------------------------
__global__ void loss_partial(const float* __restrict__ p, const float* __restrict__ y,
                             int N, float* __restrict__ part) {
    __shared__ float sp[256], sbp[256], sbn[256];
    int tid = threadIdx.x;
    float cp = 0.0f, bp = 0.0f, bn = 0.0f;
    for (int n = blockIdx.x * blockDim.x + tid; n < N; n += gridDim.x * blockDim.x) {
        float yy = y[n], pv = p[n];
        float lp = fmaxf(logf(pv), -100.0f);
        float ln = fmaxf(logf(1.0f - pv), -100.0f);
        float bce = -(yy * lp + (1.0f - yy) * ln);
        cp += yy; bp += yy * bce; bn += (1.0f - yy) * bce;
    }
    sp[tid] = cp; sbp[tid] = bp; sbn[tid] = bn;
    __syncthreads();
    for (int s = 128; s > 0; s >>= 1) {
        if (tid < s) { sp[tid] += sp[tid + s]; sbp[tid] += sbp[tid + s]; sbn[tid] += sbn[tid + s]; }
        __syncthreads();
    }
    if (tid == 0) {
        part[blockIdx.x * 3 + 0] = sp[0];
        part[blockIdx.x * 3 + 1] = sbp[0];
        part[blockIdx.x * 3 + 2] = sbn[0];
    }
}

__global__ void loss_final(const float* __restrict__ part, int nb, int N,
                           float* __restrict__ out0) {
    if (threadIdx.x == 0 && blockIdx.x == 0) {
        float cp = 0.0f, bp = 0.0f, bn = 0.0f;
        for (int i = 0; i < nb; ++i) { cp += part[i * 3]; bp += part[i * 3 + 1]; bn += part[i * 3 + 2]; }
        float neg = (float)N - cp;
        out0[0] = bp / (2.0f * cp) + bn / (2.0f * neg);
    }
}

// ---------------------------------------------------------------------------
// Host-side orchestration (all launches on `stream`, graph-capture safe).
// ---------------------------------------------------------------------------
extern "C" void kernel_launch(void* const* d_in, const int* in_sizes, int n_in,
                              void* d_out, int out_size, void* d_ws, size_t ws_size,
                              hipStream_t stream) {
    const float* x       = (const float*)d_in[0];
    const int*   ei      = (const int*)  d_in[1];   // [2,E]: src row then dst row
    const float* labels  = (const float*)d_in[2];
    const float* lin1_w  = (const float*)d_in[3];
    const float* lin1_b  = (const float*)d_in[4];
    const float* cW[3] = { (const float*)d_in[5],  (const float*)d_in[9],  (const float*)d_in[13] };
    const float* cU[3] = { (const float*)d_in[6],  (const float*)d_in[10], (const float*)d_in[14] };
    const float* cC[3] = { (const float*)d_in[7],  (const float*)d_in[11], (const float*)d_in[15] };
    const float* cB[3] = { (const float*)d_in[8],  (const float*)d_in[12], (const float*)d_in[16] };
    const float* lin2_w  = (const float*)d_in[17];
    const float* lin2_b  = (const float*)d_in[18];
    const float* lin3_w  = (const float*)d_in[19];
    const float* lin3_b  = (const float*)d_in[20];

    const int N = in_sizes[2];                 // labels: [N,1]
    const int E = in_sizes[1] / 2;             // edge_index: [2,E]
    const int Npad = (N + 15) & ~15;

    // Workspace layout (float units); everything re-initialized each call.
    float* ws   = (float*)d_ws;
    float* F0   = ws;                                   // [N,128] feature ping
    float* F1   = F0 + (size_t)N * 128;                 // [N,128] feature pong
    float* Mb   = F1 + (size_t)N * 128;                 // [N,1024] M-buffer / h5
    float* Ab   = Mb + (size_t)N * 1024;                // [N,8] attention pre-logits
    float* cnt  = Ab + (size_t)N * 8;                   // [N] degree (+1 self loop)
    float* part = cnt + N;                              // [256*3] loss partials
    _Float16* Ah = (_Float16*)(part + 1024);            // [Npad, 128] packed A frags
    _Float16* Bh = Ah + (size_t)Npad * 128;             // [128, 1024] packed B frags
    float* pout = (float*)d_out + 1;                    // p at d_out[1..N]

    const int T = 256;

    // degree (self loop counted once)
    fill_f32<<<(N + T - 1) / T, T, 0, stream>>>(cnt, 1.0f, N);
    degree_kernel<<<(E + T - 1) / T, T, 0, stream>>>(ei + E, E, cnt);

    // lin1: x[N,3] -> F0[N,16], ReLU
    lin1_kernel<<<(N + T - 1) / T, T, 0, stream>>>(x, lin1_w, lin1_b, F0, N);

    // One dense layer through the WMMA path.
    auto run_gemm = [&](const float* Afm, int cin, const float* Bfm, int kout,
                        float* Cout, const float* bias, int relu) {
        int nchunks = (cin + 31) / 32;
        long long na = (long long)Npad * nchunks * 32;
        pack_a_f16<<<(int)((na + T - 1) / T), T, 0, stream>>>(Afm, Ah, N, cin, Npad, nchunks);
        long long nbp = (long long)nchunks * kout * 32;
        pack_b_f16<<<(int)((nbp + T - 1) / T), T, 0, stream>>>(Bfm, Bh, cin, kout, nchunks);
        int tiles = (Npad / 16) * (kout / 16);
        gemm_wmma_packed<<<(tiles + 7) / 8, T, 0, stream>>>(Ah, Bh, Cout, bias,
                                                            N, Npad, nchunks, kout, relu);
    };

    // FeaStConv layers
    const int cins[3]  = {16, 32, 64};
    const int couts[3] = {32, 64, 128};
    float* bufs[2] = {F0, F1};
    for (int L = 0; L < 3; ++L) {
        const float* hin = bufs[L & 1];
        float* hout      = bufs[(L + 1) & 1];
        const int cin = cins[L], cout = couts[L], kout = 8 * cout;

        attn_pre<<<(N + T - 1) / T, T, 0, stream>>>(hin, cU[L], Ab, N, cin);
        run_gemm(hin, cin, cW[L], kout, Mb, nullptr, 0);

        long long nc = (long long)N * cout;
        fill_f32<<<(int)((nc + T - 1) / T), T, 0, stream>>>(hout, 0.0f, nc);

        int ET = E + N;  // edges + self loops, one wave32 each
        feast_edge_kernel<<<(ET + 7) / 8, T, 0, stream>>>(ei, E, N, Ab, cC[L], Mb, hout, cout);
        finish_kernel<<<(int)((nc + T - 1) / T), T, 0, stream>>>(hout, cnt, cB[L], N, cout);
    }

    // lin2: F1[N,128] @ [128,1024] + bias, ReLU -> Mb (reused as h5)
    run_gemm(F1, 128, lin2_w, 1024, Mb, lin2_b, 1);

    // lin3 + sigmoid -> p (d_out[1..N])
    lin3_sigmoid<<<(N + 7) / 8, T, 0, stream>>>(Mb, lin3_w, lin3_b, pout, N);

    // class-balanced BCE -> d_out[0]
    const int NB = 256;
    loss_partial<<<NB, T, 0, stream>>>(pout, labels, N, part);
    loss_final<<<1, 32, 0, stream>>>(part, NB, N, (float*)d_out);
}
